// NpiLstm_90469191122959
// MI455X (gfx1250) — matrix operand order repaired and checked by reading, not compiled, not run on hardware
//
#include <hip/hip_runtime.h>

typedef __attribute__((ext_vector_type(8)))  float  v8f;
typedef __attribute__((ext_vector_type(8)))  __bf16 v8bf;
typedef __attribute__((ext_vector_type(16))) __bf16 v16bf;

#define T_STEPS 512
#define B_SZ    128
#define H_SZ    1024
#define K_DIM   1088          // 64 input cols + 1024 hidden cols
#define NWG     32            // persistent workgroups
#define J_PW    32            // hidden units per WG
#define NCOL    128           // 4 gates * J_PW weight columns per WG
#define COLSTR  1096          // bf16 elems per LDS weight column (1088 + 8 pad -> 2192B, 16B aligned)
#define INPROW  1088          // elements per row of the activation buffer
#define BUFELEMS (B_SZ*INPROW)
#define KB_CNT  34            // 1088 / 32

#define SMEM_WT_BYTES (NCOL*COLSTR*2)        // 280576
#define OFF_BIAS  (SMEM_WT_BYTES)            // f32[128]
#define OFF_WFB   (OFF_BIAS + 512)           // f32[128]  W_ih[:,63] (feedback col)
#define OFF_WLIN  (OFF_WFB + 512)            // f32[32]
#define OFF_LOUT  (OFF_WLIN + 128)           // f32[128]  per-WG output partials
#define SMEM_TOTAL (OFF_LOUT + 512)          // 282240 bytes (< 320KB WGP LDS)

__device__ __forceinline__ float sigf(float x) { return 1.0f / (1.0f + __expf(-x)); }

__device__ __forceinline__ void grid_barrier(unsigned* cnt, unsigned phase, unsigned nwg) {
  __syncthreads();
  if (threadIdx.x == 0) {
    __threadfence();
    __hip_atomic_fetch_add(cnt, 1u, __ATOMIC_RELEASE, __HIP_MEMORY_SCOPE_AGENT);
    while (__hip_atomic_load(cnt, __ATOMIC_ACQUIRE, __HIP_MEMORY_SCOPE_AGENT) < phase * nwg) {
      __builtin_amdgcn_s_sleep(8);
    }
  }
  __syncthreads();
}

__global__ void lstm_prep(float* __restrict__ outacc, const float* __restrict__ b_lin,
                          unsigned* __restrict__ cnt) {
  int g = blockIdx.x * blockDim.x + threadIdx.x;
  if (g == 0) *cnt = 0u;
  if (g < T_STEPS * B_SZ) {
    int t = g >> 7;
    outacc[g] = (t == 0) ? 1e-9f : b_lin[0];   // row0 = INIT_OUT; rows t>=1 seeded with linear bias
  }
}

__global__ __launch_bounds__(256)
void lstm_persistent(const float* __restrict__ x,      // (T,B,63) f32
                     const float* __restrict__ W_ih,   // (4096,64)
                     const float* __restrict__ W_hh,   // (4096,1024)
                     const float* __restrict__ b_ih,
                     const float* __restrict__ b_hh,
                     const float* __restrict__ W_lin,  // (1,1024)
                     float* __restrict__ d_out,        // (T,B)
                     __bf16* __restrict__ inp,         // 2 * B * 1088 activation double buffer
                     float* __restrict__ outacc,       // (T,B) output accumulator
                     unsigned* __restrict__ cnt)
{
  extern __shared__ char smem[];
  __bf16* wt   = (__bf16*)smem;
  float* biasL = (float*)(smem + OFF_BIAS);
  float* wfbL  = (float*)(smem + OFF_WFB);
  float* wlinL = (float*)(smem + OFF_WLIN);
  float* loutL = (float*)(smem + OFF_LOUT);

  const int tid  = threadIdx.x;
  const int wg   = blockIdx.x;
  const int gtid = wg * 256 + tid;

  // ---- prologue: weight slice -> LDS (bf16, column-major, padded) ----
  for (int idx = tid; idx < NCOL * K_DIM; idx += 256) {
    int col = idx / K_DIM, k = idx - col * K_DIM;
    int gate = col >> 5, jl = col & 31;
    int row = gate * H_SZ + wg * J_PW + jl;        // row of stacked (4H, *) weights
    float v = (k < 64) ? W_ih[row * 64 + k] : W_hh[row * H_SZ + (k - 64)];
    wt[col * COLSTR + k] = (__bf16)v;
  }
  if (tid < NCOL) {
    int gate = tid >> 5, jl = tid & 31;
    int row = gate * H_SZ + wg * J_PW + jl;
    biasL[tid] = b_ih[row] + b_hh[row];
    wfbL[tid]  = W_ih[row * 64 + 63];              // feedback-channel weight (kept f32)
    loutL[tid] = 0.0f;
  }
  if (tid < J_PW) wlinL[tid] = W_lin[wg * J_PW + tid];

  // ---- prefill activation buffer for step 1 ----
  if (gtid < B_SZ * 63) {
    int b = gtid / 63, k2 = gtid - b * 63;
    inp[BUFELEMS + b * INPROW + k2] = (__bf16)x[1 * B_SZ * 63 + gtid];
  }
  for (int i = gtid; i < B_SZ * H_SZ; i += NWG * 256) {
    int b = i >> 10, j = i & 1023;
    inp[BUFELEMS + b * INPROW + 64 + j] = (__bf16)0.0f;   // h_0 = 0
  }
  if (gtid < B_SZ) {                                      // feedback column stays 0 forever
    inp[gtid * INPROW + 63]            = (__bf16)0.0f;
    inp[BUFELEMS + gtid * INPROW + 63] = (__bf16)0.0f;
  }

  unsigned phase = 1;
  grid_barrier(cnt, phase++, NWG);

  // ---- per-wave tile mapping ----
  const int w  = tid >> 5;      // wave 0..7
  const int l  = tid & 31;
  const int ln = l & 15;        // N within tile / A row low
  const int hi = l >> 4;        // K-half select
  const int p  = w & 3;         // batch-row pair: rows [32p, 32p+32)
  const int ut = w >> 2;        // unit tile: hidden units [16ut, 16ut+16) of this WG's slice

  const __bf16* bcol[4];
  float biasv[4], wfbv[4];
#pragma unroll
  for (int g = 0; g < 4; ++g) {
    int cg = g * 32 + ut * 16 + ln;
    bcol[g]  = wt + cg * COLSTR;
    biasv[g] = biasL[cg];
    wfbv[g]  = wfbL[cg];
  }
  const float wlinv = wlinL[ut * 16 + ln];
  const int   hcol  = 64 + wg * J_PW + ut * 16 + ln;

  v8f creg[2] = {};   // cell state, f32, lives in registers for all 511 steps

  for (int t = 1; t < T_STEPS; ++t) {
    const __bf16* bufc = inp + (size_t)(t & 1) * BUFELEMS;
    __bf16*       bufn = inp + (size_t)((t + 1) & 1) * BUFELEMS;
    const __bf16* arow[2];
    arow[0] = bufc + (p * 32 + ln) * INPROW;
    arow[1] = arow[0] + 16 * INPROW;

    // init accumulators with bias
    v8f acc[2][4];
#pragma unroll
    for (int mti = 0; mti < 2; ++mti)
#pragma unroll
      for (int g = 0; g < 4; ++g) {
        v8f a;
#pragma unroll
        for (int r = 0; r < 8; ++r) a[r] = biasv[g];
        acc[mti][g] = a;
      }

    // analytic feedback: gates += prev_out[b] * W_ih[:,63]  (f32, exact)
    const float* po = outacc + (size_t)(t - 1) * B_SZ;
#pragma unroll
    for (int mti = 0; mti < 2; ++mti) {
      int bbase = p * 32 + mti * 16 + hi * 8;
#pragma unroll
      for (int r = 0; r < 8; ++r) {
        float pv = po[bbase + r];
#pragma unroll
        for (int g = 0; g < 4; ++g) acc[mti][g][r] += pv * wfbv[g];
      }
    }

    // GEMM: (128 x 1088) @ (1088 x 128-slice) with v_wmma_f32_16x16x32_bf16
    for (int kb = 0; kb < KB_CNT; ++kb) {
      union AF { v16bf v; struct { v8bf lo, hi; } s; } afr[2], bfr[4];
      const int ka = kb * 32 + hi * 8;     // A: lanes0-15 K{0-7,16-23}, lanes16-31 K{8-15,24-31}
#pragma unroll
      for (int mti = 0; mti < 2; ++mti) {
        afr[mti].s.lo = *(const v8bf*)(arow[mti] + ka);
        afr[mti].s.hi = *(const v8bf*)(arow[mti] + ka + 16);
      }
      const int kbcol = kb * 32 + hi * 8;  // B: lanes0-15 K{0-7,16-23}, lanes16-31 K{8-15,24-31}
#pragma unroll
      for (int g = 0; g < 4; ++g) {
        bfr[g].s.lo = *(const v8bf*)(bcol[g] + kbcol);
        bfr[g].s.hi = *(const v8bf*)(bcol[g] + kbcol + 16);
      }
#pragma unroll
      for (int mti = 0; mti < 2; ++mti)
#pragma unroll
        for (int g = 0; g < 4; ++g)
          acc[mti][g] = __builtin_amdgcn_wmma_f32_16x16x32_bf16(
              false, afr[mti].v, false, bfr[g].v, (short)0, acc[mti][g], false, false);
    }

    // LSTM cell (f32), h write-back (bf16), output projection partials
#pragma unroll
    for (int mti = 0; mti < 2; ++mti) {
      int bbase = p * 32 + mti * 16 + hi * 8;
#pragma unroll
      for (int r = 0; r < 8; ++r) {
        float iv = sigf(acc[mti][0][r]);
        float fv = sigf(acc[mti][1][r]);
        float gv = tanhf(acc[mti][2][r]);
        float ov = sigf(acc[mti][3][r]);
        float cc = fv * creg[mti][r] + iv * gv;
        creg[mti][r] = cc;
        float hh = ov * tanhf(cc);
        bufn[(bbase + r) * INPROW + hcol] = (__bf16)hh;
        float part = hh * wlinv;
#pragma unroll
        for (int m = 1; m < 16; m <<= 1) part += __shfl_xor(part, m, 16);
        if (ln == 0) atomicAdd(&loutL[bbase + r], part);   // ds_add_f32
      }
    }
    __syncthreads();
    if (tid < B_SZ) {
      atomicAdd(&outacc[(size_t)t * B_SZ + tid], loutL[tid]);  // global_atomic_add_f32
      loutL[tid] = 0.0f;
    }
    // stage x_{t+1} (f32 -> bf16) into next buffer
    if (t + 1 < T_STEPS && gtid < B_SZ * 63) {
      int b = gtid / 63, k2 = gtid - b * 63;
      bufn[b * INPROW + k2] = (__bf16)x[(size_t)(t + 1) * B_SZ * 63 + gtid];
    }
    grid_barrier(cnt, phase++, NWG);
  }

  // final: outacc -> d_out
  for (int i = gtid; i < T_STEPS * B_SZ; i += NWG * 256) d_out[i] = outacc[i];
}

extern "C" void kernel_launch(void* const* d_in, const int* in_sizes, int n_in,
                              void* d_out, int out_size, void* d_ws, size_t ws_size,
                              hipStream_t stream) {
  (void)in_sizes; (void)n_in; (void)out_size; (void)ws_size;
  const float* x     = (const float*)d_in[0];
  const float* W_ih  = (const float*)d_in[1];
  const float* W_hh  = (const float*)d_in[2];
  const float* b_ih  = (const float*)d_in[3];
  const float* b_hh  = (const float*)d_in[4];
  const float* W_lin = (const float*)d_in[5];
  const float* b_lin = (const float*)d_in[6];

  char* ws = (char*)d_ws;
  __bf16*  inp    = (__bf16*)ws;                              // 557056 B
  float*   outacc = (float*)(ws + 2u * BUFELEMS * 2u);        // 262144 B
  unsigned* cnt   = (unsigned*)(ws + 2u * BUFELEMS * 2u + (size_t)T_STEPS * B_SZ * 4u);

  (void)hipFuncSetAttribute((const void*)lstm_persistent,
                            hipFuncAttributeMaxDynamicSharedMemorySize, SMEM_TOTAL);

  lstm_prep<<<(T_STEPS * B_SZ + 255) / 256, 256, 0, stream>>>(outacc, b_lin, cnt);
  lstm_persistent<<<NWG, 256, SMEM_TOTAL, stream>>>(
      x, W_ih, W_hh, b_ih, b_hh, W_lin, (float*)d_out, inp, outacc, cnt);
}
